// SRNNCell_87660282511868
// MI455X (gfx1250) — matrix operand order repaired and checked by reading, not compiled, not run on hardware
//
#include <hip/hip_runtime.h>
#include <hip/hip_bf16.h>

// ---------------- problem constants ----------------
#define NN      2048
#define NE      1024
#define NA      3
#define BSZ     512
#define INS     512
#define SS      10240            // state row stride (floats)
#define UNFOLDS 6
#define DTC     (0.04f / 6.0f)

// ---------------- GEMM tiling ----------------
// Block: 128 threads = 4 waves, tile 64(M) x 128(N).
// Waves 2x2; wave tile 32(M) x 64(N) = 2x4 WMMA accumulators.
// Per K-step(32) per wave: 2 A-frags + 4 B-frags (12 ds_load_b128) -> 8 WMMAs.
#define TM 64
#define TN 128
#define KB 32
#define LSTR 40                  // LDS row stride in halfs (80B, conflict-free)

typedef __attribute__((ext_vector_type(16))) _Float16 v16h;
typedef __attribute__((ext_vector_type(8)))  _Float16 v8h;
typedef __attribute__((ext_vector_type(8)))  float    v8f;

__device__ __forceinline__ float softplusf(float x) {
    return (x > 20.f) ? x : log1pf(expf(x));
}

// piecewise sigmoid, S_a=0.9, S_c=0 -> a=0.45, k=5 (matches reference where-chain order)
__device__ __forceinline__ float psig(float x) {
    const float a = 0.45f, k = 5.f;
    const float x1 = a - 1.f, x2 = -a, x3 = a, x4 = 1.f - a;
    float qr  = k * (x - x1) * (x - x1);
    float lin = x + 0.5f;
    float qc  = 1.f - k * (x - x4) * (x - x4);
    float o = (x < x1) ? 0.f : qr;
    o = (x >= x2) ? lin : o;
    o = (x > x3)  ? qc  : o;
    o = (x > x4)  ? 1.f : o;
    return o;
}

// ---------------- param precompute (1 thread) ----------------
// P[0]=g_x  P[1..3]=g_aE  P[4..6]=g_aI  P[7..9]=c_E  P[10..12]=c_I
// P[13..15]=c0_E  P[16..18]=c0_I  P[19]=dt/tau_rec_E P[20]=dt/tau_rel_E
// P[21]=dt/tau_rec_I P[22]=dt/tau_rel_I
__global__ void compute_params(const float* ltd,
                               const float* laElo, const float* laEhi,
                               const float* lcE, const float* c0E,
                               const float* laIlo, const float* laIhi,
                               const float* lcI, const float* c0I,
                               const float* lrecE, const float* lrelE,
                               const float* lrecI, const float* lrelI,
                               float* P) {
    if (threadIdx.x != 0 || blockIdx.x != 0) return;
    P[0] = DTC / softplusf(ltd[0]);
    float loE = softplusf(laElo[0]), hiE = softplusf(laEhi[0]);
    float loI = softplusf(laIlo[0]), hiI = softplusf(laIhi[0]);
    for (int j = 0; j < NA; ++j) {
        float tj = 0.5f * (float)j;                 // linspace(0,1,3)
        P[1 + j]  = DTC / (loE + (hiE - loE) * tj);
        P[4 + j]  = DTC / (loI + (hiI - loI) * tj);
        P[7 + j]  = softplusf(lcE[j]);
        P[10 + j] = softplusf(lcI[j]);
        P[13 + j] = c0E[j];
        P[16 + j] = c0I[j];
    }
    P[19] = DTC / softplusf(lrecE[0]);
    P[20] = DTC / softplusf(lrelE[0]);
    P[21] = DTC / softplusf(lrecI[0]);
    P[22] = DTC / softplusf(lrelI[0]);
}

// ---------------- f32 -> f16 packers ----------------
__global__ void pack_f16(const float* __restrict__ src, _Float16* __restrict__ dst, int n) {
    int t = blockIdx.x * blockDim.x + threadIdx.x;
    if (t < n) dst[t] = (_Float16)src[t];
}

__global__ void pack_weff(const float* __restrict__ Wr, const float* __restrict__ mask,
                          _Float16* __restrict__ dst) {
    int t = blockIdx.x * blockDim.x + threadIdx.x;
    if (t < NN * NN) {
        int k = t & (NN - 1);
        float s = (k < NE) ? 1.f : -1.f;
        dst[t] = (_Float16)(softplusf(Wr[t]) * s * mask[t]);
    }
}

// ---------------- elementwise unfold step ----------------
// Computes r, stores r_dep(f16), updates a (NA) and b in place in the state.
__global__ void ew_step(float* __restrict__ st, _Float16* __restrict__ rdep,
                        const float* __restrict__ P) {
    int t = blockIdx.x * blockDim.x + threadIdx.x;
    if (t >= BSZ * NN) return;
    int b = t >> 11;
    int n = t & (NN - 1);
    float* row = st + (size_t)b * SS;
    bool isE = (n < NE);
    int  e   = isE ? n : (n - NE);
    float* ap = row + (isE ? 0 : NE * NA) + e * NA;
    float* bp = row + (isE ? 2 * NE * NA : 2 * NE * NA + NE) + e;
    float  x  = row[2 * NE * NA + 2 * NE + n];           // x block at 8192 + n

    const float* cc = P + (isE ? 7 : 10);
    const float* ga = P + (isE ? 1 : 4);
    const float* c0 = P + (isE ? 13 : 16);
    float a0 = ap[0], a1 = ap[1], a2 = ap[2];
    float xeff = x - (a0 * cc[0] + a1 * cc[1] + a2 * cc[2]);
    float r = psig(xeff);
    float bb = *bp;
    rdep[t] = (_Float16)(bb * r);
    ap[0] = (a0 + ga[0] * (r + c0[0])) / (1.f + ga[0]);
    ap[1] = (a1 + ga[1] * (r + c0[1])) / (1.f + ga[1]);
    ap[2] = (a2 + ga[2] * (r + c0[2])) / (1.f + ga[2]);
    float drec = P[isE ? 19 : 21], drel = P[isE ? 20 : 22];
    *bp = (bb + drec) / (1.f + drec + r * drel);
}

// ---------------- WMMA GEMM ----------------
// C[m,n] = sum_k A[m,k] * W[n,k]   (A: M x K f16, W: N x K f16)
// MODE 0: Out(I_ext, ld=N)[m,n] = C + Csrc[n]            (Csrc = a_0)
// MODE 1: x[m,n] = (x + g_x*(C + Csrc[m,n]))/(1+g_x)     (Csrc = I_ext, Out = state base)
template <int MODE>
__global__ void __launch_bounds__(128)
gemm_wmma(const _Float16* __restrict__ A, int lda,
          const _Float16* __restrict__ W, int ldw,
          int Nc, int K,
          const float* __restrict__ Csrc,
          float* __restrict__ Out,
          const float* __restrict__ P) {
    __shared__ __align__(16) _Float16 As[TM * LSTR];
    __shared__ __align__(16) _Float16 Bs[TN * LSTR];

    const int M0 = blockIdx.y * TM;
    const int N0 = blockIdx.x * TN;
    const int tid  = threadIdx.x;
    const int lane = tid & 31;
    const int wv   = tid >> 5;               // 4 waves
    const int waveM = (wv & 1) * 32;         // 2 waves across M (64)
    const int waveN = (wv >> 1) * 64;        // 2 waves across N (128)

    const int mrow = (lane >= 16) ? 8 : 0;   // C/D layout: lanes 16-31 hold M+8
    const int ncol = lane & 15;

    v8f acc[2][4];
    #pragma unroll
    for (int mt = 0; mt < 2; ++mt)
        #pragma unroll
        for (int nt = 0; nt < 4; ++nt) {
            v8f c{};
            if (MODE == 1) {
                const float* base = Csrc + (size_t)(M0 + waveM + mt * 16 + mrow) * Nc
                                         + (N0 + waveN + nt * 16 + ncol);
                #pragma unroll
                for (int r = 0; r < 8; ++r) c[r] = base[(size_t)r * Nc];
            }
            acc[mt][nt] = c;
        }

    for (int kk = 0; kk < K; kk += KB) {
        __syncthreads();
        // stage A tile 64x32 (128 units of 16 halfs): one unit per thread
        {
            int row = tid >> 1, seg = (tid & 1) * 16;
            const float4* s = reinterpret_cast<const float4*>(A + (size_t)(M0 + row) * lda + kk + seg);
            float4* d = reinterpret_cast<float4*>(&As[row * LSTR + seg]);
            d[0] = s[0]; d[1] = s[1];
        }
        // stage W tile 128x32 (256 units): two units per thread
        {
            int row = tid >> 1, seg = (tid & 1) * 16;
            const float4* s0 = reinterpret_cast<const float4*>(W + (size_t)(N0 + row) * ldw + kk + seg);
            float4* d0 = reinterpret_cast<float4*>(&Bs[row * LSTR + seg]);
            d0[0] = s0[0]; d0[1] = s0[1];
            const float4* s1 = reinterpret_cast<const float4*>(W + (size_t)(N0 + 64 + row) * ldw + kk + seg);
            float4* d1 = reinterpret_cast<float4*>(&Bs[(64 + row) * LSTR + seg]);
            d1[0] = s1[0]; d1[1] = s1[1];
        }
        __syncthreads();

        // A fragments (16x32): lanes 0-15 -> K{0..7,16..23}, lanes 16-31 -> K{8..15,24..31}
        const int am = lane & 15;
        const int ak = (lane >= 16) ? 8 : 0;
        v16h afrag[2], bfrag[4];
        #pragma unroll
        for (int mt = 0; mt < 2; ++mt) {
            const _Float16* rp = &As[(waveM + mt * 16 + am) * LSTR];
            v8h lo = *reinterpret_cast<const v8h*>(rp + ak);
            v8h hi = *reinterpret_cast<const v8h*>(rp + ak + 16);
            v16h f;
            #pragma unroll
            for (int i = 0; i < 8; ++i) { f[i] = lo[i]; f[8 + i] = hi[i]; }
            afrag[mt] = f;
        }
        // B fragments (32x16): lanes 0-15 -> K{0..7,16..23} of col n, lanes 16-31 -> K{8..15,24..31}
        const int bk = (lane >= 16) ? 8 : 0;
        #pragma unroll
        for (int nt = 0; nt < 4; ++nt) {
            const _Float16* rp = &Bs[(waveN + nt * 16 + ncol) * LSTR + bk];
            v8h lo = *reinterpret_cast<const v8h*>(rp);
            v8h hi = *reinterpret_cast<const v8h*>(rp + 16);
            v16h f;
            #pragma unroll
            for (int i = 0; i < 8; ++i) { f[i] = lo[i]; f[8 + i] = hi[i]; }
            bfrag[nt] = f;
        }
        #pragma unroll
        for (int mt = 0; mt < 2; ++mt)
            #pragma unroll
            for (int nt = 0; nt < 4; ++nt)
                acc[mt][nt] = __builtin_amdgcn_wmma_f32_16x16x32_f16(
                    false, afrag[mt], false, bfrag[nt],
                    (short)0, acc[mt][nt], false, false);
    }

    const float gx = (MODE == 1) ? P[0] : 0.f;
    #pragma unroll
    for (int mt = 0; mt < 2; ++mt)
        #pragma unroll
        for (int nt = 0; nt < 4; ++nt) {
            int rowg = M0 + waveM + mt * 16 + mrow;
            int colg = N0 + waveN + nt * 16 + ncol;
            #pragma unroll
            for (int r = 0; r < 8; ++r) {
                float v = acc[mt][nt][r];
                if (MODE == 0) {
                    Out[(size_t)(rowg + r) * Nc + colg] = v + Csrc[colg];
                } else {
                    float* xp = Out + (size_t)(rowg + r) * SS + (2 * NE * NA + 2 * NE) + colg;
                    *xp = (*xp + gx * v) / (1.f + gx);
                }
            }
        }
}

// ---------------- workspace layout (bytes) ----------------
#define WS_P     ((size_t)0)                       // 256 B params
#define WS_WEFF  ((size_t)256)                     // 2048*2048*2 = 8388608
#define WS_WIN   (WS_WEFF + (size_t)NN * NN * 2)   // 2048*512*2  = 2097152
#define WS_U     (WS_WIN + (size_t)NN * INS * 2)   // 512*512*2   = 524288
#define WS_IEXT  (WS_U + (size_t)BSZ * INS * 2)    // 512*2048*4  = 4194304
#define WS_RDEP  (WS_IEXT + (size_t)BSZ * NN * 4)  // 512*2048*2  = 2097152
// total ~17.3 MB

extern "C" void kernel_launch(void* const* d_in, const int* in_sizes, int n_in,
                              void* d_out, int out_size, void* d_ws, size_t ws_size,
                              hipStream_t stream) {
    const float* W_raw = (const float*)d_in[0];
    const float* W_in  = (const float*)d_in[1];
    const float* mask  = (const float*)d_in[2];
    const float* a_0   = (const float*)d_in[3];
    const float* ltd   = (const float*)d_in[4];
    const float* laElo = (const float*)d_in[5];
    const float* laEhi = (const float*)d_in[6];
    const float* lcE   = (const float*)d_in[7];
    const float* c0E   = (const float*)d_in[8];
    const float* laIlo = (const float*)d_in[9];
    const float* laIhi = (const float*)d_in[10];
    const float* lcI   = (const float*)d_in[11];
    const float* c0I   = (const float*)d_in[12];
    const float* lrecE = (const float*)d_in[13];
    const float* lrelE = (const float*)d_in[14];
    const float* lrecI = (const float*)d_in[15];
    const float* lrelI = (const float*)d_in[16];
    const float* u     = (const float*)d_in[17];
    const float* state = (const float*)d_in[18];

    char* ws = (char*)d_ws;
    float*    P      = (float*)(ws + WS_P);
    _Float16* Weff16 = (_Float16*)(ws + WS_WEFF);
    _Float16* Win16  = (_Float16*)(ws + WS_WIN);
    _Float16* u16    = (_Float16*)(ws + WS_U);
    float*    Iext   = (float*)(ws + WS_IEXT);
    _Float16* rdep   = (_Float16*)(ws + WS_RDEP);
    float*    st     = (float*)d_out;

    // state -> output buffer (we update the output layout in place)
    hipMemcpyAsync(d_out, state, (size_t)BSZ * SS * sizeof(float),
                   hipMemcpyDeviceToDevice, stream);

    compute_params<<<1, 1, 0, stream>>>(ltd, laElo, laEhi, lcE, c0E,
                                        laIlo, laIhi, lcI, c0I,
                                        lrecE, lrelE, lrecI, lrelI, P);

    pack_weff<<<(NN * NN) / 256, 256, 0, stream>>>(W_raw, mask, Weff16);
    pack_f16<<<(NN * INS) / 256, 256, 0, stream>>>(W_in, Win16, NN * INS);
    pack_f16<<<(BSZ * INS) / 256, 256, 0, stream>>>(u, u16, BSZ * INS);

    // I_ext = u @ W_in^T + a_0
    dim3 grid(NN / TN, BSZ / TM);
    gemm_wmma<0><<<grid, 128, 0, stream>>>(u16, INS, Win16, INS, NN, INS,
                                           a_0, Iext, P);

    for (int s = 0; s < UNFOLDS; ++s) {
        ew_step<<<(BSZ * NN) / 256, 256, 0, stream>>>(st, rdep, P);
        gemm_wmma<1><<<grid, 128, 0, stream>>>(rdep, NN, Weff16, NN, NN, NN,
                                               Iext, st, P);
    }
}
